// GaborConv2d_1726576855968
// MI455X (gfx1250) — matrix operand order repaired and checked by reading, not compiled, not run on hardware
//
#include <hip/hip_runtime.h>
#include <hip/hip_bf16.h>
#include <math.h>

typedef __attribute__((ext_vector_type(16))) _Float16 v16h;
typedef __attribute__((ext_vector_type(8)))  _Float16 v8h;
typedef __attribute__((ext_vector_type(8)))  float    v8f;

#define HW       512
#define CH_IN    3
#define CH_OUT   64
#define KS       7
#define PADW     3
#define KTOT     147          // 3*7*7
#define KPAD     160          // 5 wmma k-steps of 32
#define KSTEPS   5
#define TILE_X   128          // pixels per block
#define TILE_W   (TILE_X + KS - 1)   // 134 unique input columns
#define COL_STRIDE 168        // halves per px row in LDS (160 used + 8 pad, 336B: 16B aligned)

// ---------------------------------------------------------------------------
// Kernel 1: build the Gabor filter bank directly in WMMA A-matrix lane layout.
// wA is indexed as t = ((mt*5 + s)*32 + lane)*16 + j  (mt = m-tile 0..3).
// For lane/j the logical K is:  K = 32*s + j + (j>=8 ? 8 : 0) + (lane>=16 ? 8 : 0)
// and the logical row is m = 16*mt + (lane & 15).
// ---------------------------------------------------------------------------
__global__ void gabor_weights_kernel(const float* __restrict__ freq,
                                     const float* __restrict__ theta,
                                     const float* __restrict__ sigma,
                                     const float* __restrict__ psi,
                                     _Float16* __restrict__ wA)
{
    int t = blockIdx.x * blockDim.x + threadIdx.x;
    if (t >= 4 * KSTEPS * 32 * 16) return;

    int j    = t & 15;
    int lane = (t >> 4) & 31;
    int s    = (t >> 9) % KSTEPS;
    int mt   = t / (KSTEPS * 512);

    int m = mt * 16 + (lane & 15);
    int K = s * 32 + j + ((j >= 8) ? 8 : 0) + ((lane >= 16) ? 8 : 0);

    float w = 0.0f;
    if (K < KTOT) {
        int i  = K / 49;
        int r  = K - i * 49;
        int ky = r / 7;
        int kx = r - ky * 7;
        // linspace(-3, 4, 7): step 7/6
        float yv = -3.0f + (float)ky * (7.0f / 6.0f);
        float xv = -3.0f + (float)kx * (7.0f / 6.0f);
        int p = m * CH_IN + i;
        float fr = freq[p], th = theta[p], sg = sigma[p], ps = psi[p];
        float ct = cosf(th), st = sinf(th);
        float rotx =  xv * ct + yv * st;
        float roty = -xv * st + yv * ct;
        float sd = sg + 0.001f;
        float g = expf(-0.5f * (rotx * rotx + roty * roty) / (sd * sd));
        g *= cosf(fr * rotx + ps);
        g /= (2.0f * 3.14159265358979323846f * sg * sg);
        w = g;
    }
    wA[t] = (_Float16)w;
}

// ---------------------------------------------------------------------------
// Kernel 2: implicit-GEMM conv. One block = 64 out channels x 128 pixels of
// one output row. 8 waves: wave%4 selects the 16-channel M-tile, wave/4
// selects which half of the 8 sub-tiles of 16 pixels it sweeps.
// im2col tile is built by zero-fill + scatter: each unique input element is
// loaded from global exactly once and broadcast to its <=7 (px,kx) LDS slots.
// ---------------------------------------------------------------------------
__global__ __launch_bounds__(256) void gabor_conv_kernel(
    const float* __restrict__ in,
    const _Float16* __restrict__ wA,
    float* __restrict__ out)
{
    __shared__ _Float16 col[TILE_X * COL_STRIDE];   // 43008 B im2col tile (f16)

    const int x0  = blockIdx.x * TILE_X;
    const int y   = blockIdx.y;
    const int n   = blockIdx.z;
    const int tid = threadIdx.x;

    // ---- phase 0: vector-zero the LDS tile (covers all padding cases) ----
    {
        uint4* cz = (uint4*)col;
        const int nvec = (TILE_X * COL_STRIDE * (int)sizeof(_Float16)) / 16; // 2688
        for (int idx = tid; idx < nvec; idx += 256)
            cz[idx] = uint4{0u, 0u, 0u, 0u};
    }
    __syncthreads();

    // ---- phase 1: scatter im2col. 21 rows x 134 cols unique input elems ----
    const float* inN = in + (size_t)n * CH_IN * HW * HW;
    for (int e = tid; e < CH_IN * KS * TILE_W; e += 256) {
        int row  = e / TILE_W;           // 0..20  (i*7 + ky)
        int xcol = e - row * TILE_W;     // 0..133
        int i    = row / KS;
        int ky   = row - i * KS;
        int yy   = y + ky - PADW;
        int xx   = x0 + xcol - PADW;
        if (yy < 0 || yy >= HW || xx < 0 || xx >= HW) continue;
        _Float16 h = (_Float16)inN[((size_t)i * HW + yy) * HW + xx];
        int kbase = row * KS;            // i*49 + ky*7
        _Float16* cb = col + kbase;
#pragma unroll
        for (int kx = 0; kx < KS; ++kx) {
            int px = xcol - kx;
            if (px >= 0 && px < TILE_X)
                cb[px * COL_STRIDE + kx] = h;
        }
    }
    __syncthreads();

    const int wave    = tid >> 5;
    const int lane    = tid & 31;
    const int mt      = wave & 3;           // which 16-channel tile
    const int subBase = (wave >> 2) * 4;    // sub-tiles 0..3 or 4..7
    const int m0      = mt * 16;
    const int koff    = (lane >= 16) ? 8 : 0;
    const int nl      = lane & 15;

    // ---- load A (weights) for this m-tile: 5 k-steps, kept in VGPRs ----
    v16h a[KSTEPS];
#pragma unroll
    for (int s = 0; s < KSTEPS; ++s)
        a[s] = *(const v16h*)(wA + (((size_t)(mt * KSTEPS + s) * 32 + lane) << 4));

#pragma unroll
    for (int sub = 0; sub < 4; ++sub) {
        const int xsub = (subBase + sub) * 16;
        const int px   = xsub + nl;
        const _Float16* crow = col + px * COL_STRIDE + koff;  // 16B aligned

        v8f acc = {};
#pragma unroll
        for (int s = 0; s < KSTEPS; ++s) {
            v8h blo = *(const v8h*)(crow + s * 32);
            v8h bhi = *(const v8h*)(crow + s * 32 + 16);
            v16h b  = __builtin_shufflevector(blo, bhi,
                        0, 1, 2, 3, 4, 5, 6, 7, 8, 9, 10, 11, 12, 13, 14, 15);
            // (neg_a, A, neg_b, B, c_mod, C, reuse_a, reuse_b)
            acc = __builtin_amdgcn_wmma_f32_16x16x32_f16(
                      false, a[s], false, b, (short)0, acc, false, false);
        }

        // ---- store D: row r holds M = r (lanes 0-15) / r+8 (lanes 16-31) ----
        const int x  = x0 + xsub + nl;
        const int ob = m0 + ((lane >= 16) ? 8 : 0);
        float* op = out + (((size_t)n * CH_OUT + ob) * HW + y) * HW + x;
#pragma unroll
        for (int r = 0; r < 8; ++r)
            op[(size_t)r * HW * HW] = acc[r];
    }
}

// ---------------------------------------------------------------------------
extern "C" void kernel_launch(void* const* d_in, const int* in_sizes, int n_in,
                              void* d_out, int out_size, void* d_ws, size_t ws_size,
                              hipStream_t stream)
{
    const float* input = (const float*)d_in[0];
    const float* freq  = (const float*)d_in[1];
    const float* theta = (const float*)d_in[2];
    const float* sigma = (const float*)d_in[3];
    const float* psi   = (const float*)d_in[4];

    _Float16* wA = (_Float16*)d_ws;          // 10240 halves = 20480 B of scratch

    gabor_weights_kernel<<<40, 256, 0, stream>>>(freq, theta, sigma, psi, wA);

    dim3 grid(HW / TILE_X, HW, 16);          // (4, 512, 16)
    gabor_conv_kernel<<<grid, 256, 0, stream>>>(input, wA, (float*)d_out);
}
